// PlantCCCEncoder_89481348645084
// MI455X (gfx1250) — matrix operand order repaired and checked by reading, not compiled, not run on hardware
//
#include <hip/hip_runtime.h>

#define N_NODES 50000
#define E_EDGES 800000
#define IN_DIM  256
#define HID     128
#define NHEAD   4
#define CH      32
#define NLAYERS 3
#define RELN    100
#define RD      16
#define EH      32

typedef __attribute__((ext_vector_type(16))) __bf16 v16bf;
typedef __attribute__((ext_vector_type(8)))  __bf16 v8bf;
typedef __attribute__((ext_vector_type(8)))  float  v8f;

// ---------- ordered-uint encoding for float atomicMax ----------
__device__ __forceinline__ unsigned fkey(float f) {
    unsigned b = __float_as_uint(f);
    return (b & 0x80000000u) ? ~b : (b | 0x80000000u);
}
__device__ __forceinline__ float fdecode(unsigned k) {
    unsigned b = (k & 0x80000000u) ? (k & 0x7FFFFFFFu) : ~k;
    return __uint_as_float(b);
}

// =================================================================
// Kernel 0: pack a K x Nc f32 weight matrix into WMMA B-fragment
// layout as bf16:  Bp[((ct*(K/32)+kc)*32 + lane)*16 + j]
//   col = ct*16 + (lane&15),  k = kc*32 + (lane>>4)*8 + (j&7) + (j>>3)*16
// so each lane's fragment per K-step is one contiguous 32B chunk.
// =================================================================
__global__ void k_pack_b(const float* __restrict__ B, __bf16* __restrict__ Bp,
                         int K, int Nc)
{
    int idx = blockIdx.x * blockDim.x + threadIdx.x;
    int total = K * Nc;
    if (idx >= total) return;
    int j    = idx & 15;
    int lane = (idx >> 4) & 31;
    int rem  = idx >> 9;
    int kcN  = K >> 5;
    int kc   = rem % kcN;
    int ct   = rem / kcN;
    int col  = (ct << 4) + (lane & 15);
    int k    = (kc << 5) + ((lane >> 4) << 3) + (j & 7) + ((j >> 3) << 4);
    Bp[idx] = (__bf16)B[(size_t)k * Nc + col];
}

// =================================================================
// Kernel 1: per-edge feature MLP  ->  edge_feat[E][16]
// =================================================================
__global__ void k_edge_feat(const float* __restrict__ edge_attr,
                            const float* __restrict__ dW1, const float* __restrict__ db1,
                            const float* __restrict__ dW2, const float* __restrict__ db2,
                            const float* __restrict__ cW1, const float* __restrict__ cb1,
                            const float* __restrict__ cW2, const float* __restrict__ cb2,
                            const float* __restrict__ rel_emb,
                            const float* __restrict__ fW1, const float* __restrict__ fb1,
                            const float* __restrict__ flg, const float* __restrict__ flb,
                            const float* __restrict__ fW2, const float* __restrict__ fb2,
                            float* __restrict__ efeat)
{
    __shared__ float sm[4944];
    {
        const float* srcs[15] = {dW1, db1, dW2, db2, cW1, cb1, cW2, cb2,
                                 rel_emb, fW1, fb1, flg, flb, fW2, fb2};
        const int sz[15]  = {32, 32, 512, 16, 32, 32, 512, 16, 1600, 1536, 32, 32, 32, 512, 16};
        const int off[15] = {0, 32, 64, 576, 592, 624, 656, 1168, 1184, 2784, 4320, 4352, 4384, 4416, 4928};
        for (int a = 0; a < 15; ++a)
            for (int i = threadIdx.x; i < sz[a]; i += blockDim.x)
                sm[off[a] + i] = srcs[a][i];
    }
    __syncthreads();

    int e = blockIdx.x * blockDim.x + threadIdx.x;
    if (e >= E_EDGES) return;

    float dval = edge_attr[(size_t)e * 3 + 0];
    float cval = edge_attr[(size_t)e * 3 + 1];
    int rel = (int)edge_attr[(size_t)e * 3 + 2];
    rel = rel < 0 ? 0 : (rel > RELN - 1 ? RELN - 1 : rel);

    float comb[48];
    {
        float h[32];
        #pragma unroll
        for (int j = 0; j < 32; ++j) { float v = dval * sm[0 + j] + sm[32 + j]; h[j] = v > 0.f ? v : 0.f; }
        #pragma unroll
        for (int k = 0; k < 16; ++k) {
            float s = sm[576 + k];
            #pragma unroll
            for (int j = 0; j < 32; ++j) s += h[j] * sm[64 + j * 16 + k];
            comb[k] = s;
        }
    }
    {
        float h[32];
        #pragma unroll
        for (int j = 0; j < 32; ++j) { float v = cval * sm[592 + j] + sm[624 + j]; h[j] = v > 0.f ? v : 0.f; }
        #pragma unroll
        for (int k = 0; k < 16; ++k) {
            float s = sm[1168 + k];
            #pragma unroll
            for (int j = 0; j < 32; ++j) s += h[j] * sm[656 + j * 16 + k];
            comb[16 + k] = s;
        }
    }
    #pragma unroll
    for (int k = 0; k < 16; ++k) comb[32 + k] = sm[1184 + rel * 16 + k];

    float fh[32];
    #pragma unroll
    for (int j = 0; j < 32; ++j) fh[j] = sm[4320 + j];
    #pragma unroll
    for (int t = 0; t < 48; ++t) {
        float c = comb[t];
        #pragma unroll
        for (int j = 0; j < 32; ++j) fh[j] += c * sm[2784 + t * 32 + j];
    }
    float mean = 0.f, sq = 0.f;
    #pragma unroll
    for (int j = 0; j < 32; ++j) mean += fh[j];
    mean *= (1.0f / 32.0f);
    #pragma unroll
    for (int j = 0; j < 32; ++j) { float d = fh[j] - mean; sq += d * d; }
    float rs = rsqrtf(sq * (1.0f / 32.0f) + 1e-5f);
    #pragma unroll
    for (int j = 0; j < 32; ++j) {
        float y = (fh[j] - mean) * rs * sm[4352 + j] + sm[4384 + j];
        fh[j] = y > 0.f ? y : 0.f;
    }
    #pragma unroll
    for (int k = 0; k < 16; ++k) {
        float s = sm[4928 + k];
        #pragma unroll
        for (int j = 0; j < 32; ++j) s += fh[j] * sm[4416 + j * 16 + k];
        efeat[(size_t)e * 16 + k] = s;
    }
}

// =================================================================
// Kernel 2: input LayerNorm -> bf16  (one wave per node row of 256)
// =================================================================
__global__ void k_ln_in(const float* __restrict__ x,
                        const float* __restrict__ g, const float* __restrict__ b,
                        __bf16* __restrict__ lnx)
{
    const int lane = threadIdx.x & 31;
    const int n = blockIdx.x * (blockDim.x >> 5) + (threadIdx.x >> 5);
    if (n >= N_NODES) return;
    const float* row = x + (size_t)n * IN_DIM + lane * 8;
    float4 v0 = *(const float4*)(row);
    float4 v1 = *(const float4*)(row + 4);
    float s = v0.x + v0.y + v0.z + v0.w + v1.x + v1.y + v1.z + v1.w;
    float ss = v0.x*v0.x + v0.y*v0.y + v0.z*v0.z + v0.w*v0.w
             + v1.x*v1.x + v1.y*v1.y + v1.z*v1.z + v1.w*v1.w;
    #pragma unroll
    for (int m = 1; m < 32; m <<= 1) { s += __shfl_xor(s, m, 32); ss += __shfl_xor(ss, m, 32); }
    float mean = s * (1.0f / IN_DIM);
    float var = ss * (1.0f / IN_DIM) - mean * mean;
    float rs = rsqrtf(var + 1e-5f);
    float vv[8] = {v0.x, v0.y, v0.z, v0.w, v1.x, v1.y, v1.z, v1.w};
    #pragma unroll
    for (int j = 0; j < 8; ++j) {
        int c = lane * 8 + j;
        lnx[(size_t)n * IN_DIM + c] = (__bf16)((vv[j] - mean) * rs * g[c] + b[c]);
    }
}

// =================================================================
// Kernel 3: WMMA bf16 GEMM with pre-packed B fragments.
//   C[M x Nc] = A[M x K] @ B[K x Nc] (+bias)
//   A: bf16 row-major. Bp: fragment-layout bf16 (see k_pack_b).
//   16 rows per block, one 16-col tile per wave, K fully unrolled.
// =================================================================
template<int K, bool BIAS, bool WBF>
__global__ void k_gemm_bf16(const __bf16* __restrict__ A,
                            const __bf16* __restrict__ Bp,
                            const float* __restrict__ bias,
                            float* __restrict__ C,
                            __bf16* __restrict__ Cbf,
                            int Nc)
{
    const int lane = threadIdx.x & 31;
    const int wave = threadIdx.x >> 5;
    const int nwave = blockDim.x >> 5;
    const int rowBase = blockIdx.x * 16;
    const int r  = lane & 15;
    const int hi = lane >> 4;
    const int row = rowBase + r;
    const __bf16* arow = A + (size_t)row * K + (hi ? 8 : 0);

    for (int ct = wave; ct < (Nc >> 4); ct += nwave) {
        const __bf16* bp = Bp + ((size_t)ct * (K / 32) * 32 + lane) * 16;
        v8f acc = {};
        #pragma unroll
        for (int kc = 0; kc < K / 32; ++kc) {
            v8bf alo = *(const v8bf*)(arow + kc * 32);        // K = koff+0..7
            v8bf ahi = *(const v8bf*)(arow + kc * 32 + 16);   // K = 16+koff+0..7
            v16bf a = __builtin_shufflevector(alo, ahi,
                        0,1,2,3,4,5,6,7,8,9,10,11,12,13,14,15);
            v16bf b = *(const v16bf*)(bp + (size_t)kc * 512); // one 32B fragment
            acc = __builtin_amdgcn_wmma_f32_16x16x32_bf16(
                      false, a, false, b, (short)0, acc, false, false);
        }
        const int col = (ct << 4) + r;
        float bv = BIAS ? bias[col] : 0.0f;
        #pragma unroll
        for (int v = 0; v < 8; ++v) {
            int m = rowBase + v + (hi ? 8 : 0);
            float val = acc[v] + bv;
            C[(size_t)m * Nc + col] = val;
            if (WBF) Cbf[(size_t)m * Nc + col] = (__bf16)val;
        }
    }
}

// =================================================================
// Kernel 4: edge attention logits (wave per edge) + segment max
// =================================================================
__global__ void k_edge_logits(const float* __restrict__ xs, const float* __restrict__ xd,
                              const float* __restrict__ efeat,
                              const float* __restrict__ Wedge,   // 16 x 128
                              const float* __restrict__ att,     // 128
                              const float* __restrict__ sscale,  // 4
                              const float* __restrict__ edge_attr,
                              const int* __restrict__ ei,
                              float* __restrict__ alpha,
                              unsigned* __restrict__ amax,
                              int apply_scale)
{
    __shared__ float sW[16 * 128];
    __shared__ float sAtt[128];
    __shared__ float sS[4];
    for (int i = threadIdx.x; i < 2048; i += blockDim.x) sW[i] = Wedge[i];
    for (int i = threadIdx.x; i < 128; i += blockDim.x) sAtt[i] = att[i];
    if (threadIdx.x < 4) sS[threadIdx.x] = sscale[threadIdx.x];
    __syncthreads();

    const int lane = threadIdx.x & 31;
    const int e = blockIdx.x * (blockDim.x >> 5) + (threadIdx.x >> 5);
    if (e >= E_EDGES) return;
    const int src = ei[e];
    const int dst = ei[E_EDGES + e];
    const int c4 = lane * 4;

    float4 vs = *(const float4*)(xs + (size_t)src * HID + c4);
    float4 vd = *(const float4*)(xd + (size_t)dst * HID + c4);

    float ef0 = 0.f, ef1 = 0.f, ef2 = 0.f, ef3 = 0.f;
    #pragma unroll
    for (int k = 0; k < 16; ++k) {
        float f = efeat[(size_t)e * 16 + k];
        const float* wr = sW + k * 128 + c4;
        ef0 += f * wr[0]; ef1 += f * wr[1]; ef2 += f * wr[2]; ef3 += f * wr[3];
    }
    float p = tanhf(vs.x + vd.x + ef0) * sAtt[c4]
            + tanhf(vs.y + vd.y + ef1) * sAtt[c4 + 1]
            + tanhf(vs.z + vd.z + ef2) * sAtt[c4 + 2]
            + tanhf(vs.w + vd.w + ef3) * sAtt[c4 + 3];
    p += __shfl_xor(p, 1, 32);
    p += __shfl_xor(p, 2, 32);
    p += __shfl_xor(p, 4, 32);   // 8-lane group == one head
    if ((lane & 7) == 0) {
        int h = lane >> 3;
        if (apply_scale) {
            float dw = edge_attr[(size_t)e * 3];
            p *= __powf(dw, sS[h]);
        }
        alpha[(size_t)e * NHEAD + h] = p;
        atomicMax(amax + (size_t)dst * NHEAD + h, fkey(p));
    }
}

// =================================================================
// Kernel 5: ea = exp(alpha - amax[dst]); segment-sum into esum
// =================================================================
__global__ void k_edge_exp(const int* __restrict__ ei,
                           float* __restrict__ alpha,
                           const unsigned* __restrict__ amax,
                           float* __restrict__ esum)
{
    int idx = blockIdx.x * blockDim.x + threadIdx.x;
    if (idx >= E_EDGES * NHEAD) return;
    int e = idx >> 2, h = idx & 3;
    int dst = ei[E_EDGES + e];
    float m = fdecode(amax[(size_t)dst * NHEAD + h]);
    float ea = __expf(alpha[idx] - m);
    alpha[idx] = ea;
    atomicAdd(esum + (size_t)dst * NHEAD + h, ea);
}

// =================================================================
// Kernel 6: msg scatter: out[dst] += xs[src] * ea/(esum[dst]+1e-8)
// =================================================================
__global__ void k_edge_msg(const int* __restrict__ ei,
                           const float* __restrict__ alpha,
                           const float* __restrict__ esum,
                           const float* __restrict__ xs,
                           float* __restrict__ out)
{
    const int lane = threadIdx.x & 31;
    const int e = blockIdx.x * (blockDim.x >> 5) + (threadIdx.x >> 5);
    if (e >= E_EDGES) return;
    const int src = ei[e];
    const int dst = ei[E_EDGES + e];
    const int c4 = lane * 4;
    const int h = lane >> 3;
    float w = alpha[(size_t)e * NHEAD + h] / (esum[(size_t)dst * NHEAD + h] + 1e-8f);
    float4 vs = *(const float4*)(xs + (size_t)src * HID + c4);
    float* o = out + (size_t)dst * HID + c4;
    atomicAdd(o + 0, vs.x * w);
    atomicAdd(o + 1, vs.y * w);
    atomicAdd(o + 2, vs.z * w);
    atomicAdd(o + 3, vs.w * w);
}

// =================================================================
// Kernel 7: x = elu(LN(out)) + x_res   (wave per node; f32 + bf16 out)
// =================================================================
__global__ void k_node_update(const float* __restrict__ out,
                              const float* __restrict__ ng, const float* __restrict__ nb,
                              const float* __restrict__ xres,
                              float* __restrict__ xnew,
                              __bf16* __restrict__ xbf)
{
    const int lane = threadIdx.x & 31;
    const int n = blockIdx.x * (blockDim.x >> 5) + (threadIdx.x >> 5);
    if (n >= N_NODES) return;
    const int c4 = lane * 4;
    float4 v = *(const float4*)(out + (size_t)n * HID + c4);
    float s = v.x + v.y + v.z + v.w;
    float ss = v.x*v.x + v.y*v.y + v.z*v.z + v.w*v.w;
    #pragma unroll
    for (int m = 1; m < 32; m <<= 1) { s += __shfl_xor(s, m, 32); ss += __shfl_xor(ss, m, 32); }
    float mean = s * (1.0f / HID);
    float var = ss * (1.0f / HID) - mean * mean;
    float rs = rsqrtf(var + 1e-5f);
    float vv[4] = {v.x, v.y, v.z, v.w};
    #pragma unroll
    for (int j = 0; j < 4; ++j) {
        int c = c4 + j;
        float y = (vv[j] - mean) * rs * ng[c] + nb[c];
        y = y > 0.f ? y : (__expf(y) - 1.0f);          // ELU(alpha=1)
        float o = y + xres[(size_t)n * HID + c];
        xnew[(size_t)n * HID + c] = o;
        xbf[(size_t)n * HID + c] = (__bf16)o;
    }
}

// =================================================================
extern "C" void kernel_launch(void* const* d_in, const int* in_sizes, int n_in,
                              void* d_out, int out_size, void* d_ws, size_t ws_size,
                              hipStream_t stream)
{
    const float* x         = (const float*)d_in[0];
    const int*   ei        = (const int*)  d_in[1];
    const float* edge_attr = (const float*)d_in[2];
    const float* in_g      = (const float*)d_in[3];
    const float* in_b      = (const float*)d_in[4];
    const float* proj_W    = (const float*)d_in[5];
    const float* proj_b    = (const float*)d_in[6];
    const float* dW1 = (const float*)d_in[7];  const float* db1 = (const float*)d_in[8];
    const float* dW2 = (const float*)d_in[9];  const float* db2 = (const float*)d_in[10];
    const float* cW1 = (const float*)d_in[11]; const float* cb1 = (const float*)d_in[12];
    const float* cW2 = (const float*)d_in[13]; const float* cb2 = (const float*)d_in[14];
    const float* rel_emb = (const float*)d_in[15];
    const float* fW1 = (const float*)d_in[16]; const float* fb1 = (const float*)d_in[17];
    const float* flg = (const float*)d_in[18]; const float* flb = (const float*)d_in[19];
    const float* fW2 = (const float*)d_in[20]; const float* fb2 = (const float*)d_in[21];
    const float* Wsrc  = (const float*)d_in[22];
    const float* Wdst  = (const float*)d_in[23];
    const float* Wedge = (const float*)d_in[24];
    const float* att    = (const float*)d_in[25];
    const float* sscale = (const float*)d_in[26];
    const float* ng     = (const float*)d_in[27];
    const float* nb     = (const float*)d_in[28];
    (void)in_sizes; (void)n_in; (void)out_size; (void)ws_size;

    // ---- workspace carve-up ----
    char* ws = (char*)d_ws;
    size_t off = 0;
    auto carve = [&](size_t bytes) { char* p = ws + off; off = (off + bytes + 255) & ~(size_t)255; return p; };
    __bf16*  lnx   = (__bf16*) carve((size_t)N_NODES * IN_DIM * 2);
    __bf16*  xbf   = (__bf16*) carve((size_t)N_NODES * HID * 2);
    float*   xbuf  = (float*)  carve((size_t)N_NODES * HID * 4);
    float*   xsb   = (float*)  carve((size_t)N_NODES * HID * 4);
    float*   xdb   = (float*)  carve((size_t)N_NODES * HID * 4);
    float*   outb  = (float*)  carve((size_t)N_NODES * HID * 4);
    float*   efeat = (float*)  carve((size_t)E_EDGES * RD * 4);
    float*   alpha = (float*)  carve((size_t)E_EDGES * NHEAD * 4);
    unsigned* amax = (unsigned*)carve((size_t)N_NODES * NHEAD * 4);
    float*   esum  = (float*)  carve((size_t)N_NODES * NHEAD * 4);
    __bf16*  bpProj = (__bf16*) carve((size_t)IN_DIM * HID * 2);
    __bf16*  bpSrc  = (__bf16*) carve((size_t)NLAYERS * HID * HID * 2);
    __bf16*  bpDst  = (__bf16*) carve((size_t)NLAYERS * HID * HID * 2);

    // ---- pack all GEMM weights into WMMA B-fragment layout ----
    k_pack_b<<<(IN_DIM * HID + 255) / 256, 256, 0, stream>>>(proj_W, bpProj, IN_DIM, HID);
    for (int i = 0; i < NLAYERS; ++i) {
        k_pack_b<<<(HID * HID + 255) / 256, 256, 0, stream>>>(
            Wsrc + (size_t)i * HID * HID, bpSrc + (size_t)i * HID * HID, HID, HID);
        k_pack_b<<<(HID * HID + 255) / 256, 256, 0, stream>>>(
            Wdst + (size_t)i * HID * HID, bpDst + (size_t)i * HID * HID, HID, HID);
    }

    // ---- edge feature MLP ----
    k_edge_feat<<<(E_EDGES + 255) / 256, 256, 0, stream>>>(
        edge_attr, dW1, db1, dW2, db2, cW1, cb1, cW2, cb2, rel_emb,
        fW1, fb1, flg, flb, fW2, fb2, efeat);

    // ---- input LN -> bf16, then proj GEMM (WMMA, K=256) ----
    k_ln_in<<<(N_NODES + 7) / 8, 256, 0, stream>>>(x, in_g, in_b, lnx);
    k_gemm_bf16<IN_DIM, true, true><<<N_NODES / 16, 256, 0, stream>>>(
        lnx, bpProj, proj_b, xbuf, xbf, HID);

    for (int i = 0; i < NLAYERS; ++i) {
        hipMemsetAsync(amax, 0, (size_t)N_NODES * NHEAD * 4, stream);  // key 0 == -inf
        hipMemsetAsync(esum, 0, (size_t)N_NODES * NHEAD * 4, stream);
        hipMemsetAsync(outb, 0, (size_t)N_NODES * HID * 4, stream);

        // xs = x @ Wsrc[i],  xd = x @ Wdst[i]   (WMMA, K=128)
        k_gemm_bf16<HID, false, false><<<N_NODES / 16, 256, 0, stream>>>(
            xbf, bpSrc + (size_t)i * HID * HID, nullptr, xsb, nullptr, HID);
        k_gemm_bf16<HID, false, false><<<N_NODES / 16, 256, 0, stream>>>(
            xbf, bpDst + (size_t)i * HID * HID, nullptr, xdb, nullptr, HID);

        k_edge_logits<<<E_EDGES / 8, 256, 0, stream>>>(
            xsb, xdb, efeat, Wedge + (size_t)i * RD * HID, att + (size_t)i * NHEAD * CH,
            sscale + (size_t)i * NHEAD, edge_attr, ei, alpha, amax,
            (i < NLAYERS - 1) ? 1 : 0);

        k_edge_exp<<<(E_EDGES * NHEAD + 255) / 256, 256, 0, stream>>>(ei, alpha, amax, esum);

        k_edge_msg<<<E_EDGES / 8, 256, 0, stream>>>(ei, alpha, esum, xsb, outb);

        float* dstF = (i == NLAYERS - 1) ? (float*)d_out : xbuf;
        k_node_update<<<(N_NODES + 7) / 8, 256, 0, stream>>>(
            outb, ng + (size_t)i * HID, nb + (size_t)i * HID, xbuf, dstF, xbf);
    }
}
